// PCT_20529943675332
// MI455X (gfx1250) — matrix operand order repaired and checked by reading, not compiled, not run on hardware
//
#include <hip/hip_runtime.h>
#include <hip/hip_bf16.h>

// ---------------------------------------------------------------------------
// PCT forward for MI455X (gfx1250, wave32). All dense GEMMs use
// v_wmma_f32_16x16x32_f16 (f32 accum, f16 operands converted on load).
// Main GEMM: block = 64x128 C tile, B tile staged transposed in LDS so the
// per-lane B fragment is two ds_load_b128; A loaded via global b128s.
// ---------------------------------------------------------------------------

typedef __attribute__((ext_vector_type(16))) _Float16 v16h;
typedef __attribute__((ext_vector_type(8)))  _Float16 v8h;
typedef __attribute__((ext_vector_type(8)))  float    v8f;

static constexpr int B_ = 16;

// ==================== LDS-staged batched WMMA GEMM (main) ===================
// C[M x N] = A[M x K] * B[K x N], row-major logical. A is f32.
// Requires M % 64 == 0, N % 128 == 0, K % 32 == 0.
// TRA: A stored as raw[k*M + m] (raw K x M row-major), e.g. q^T @ k.
template<typename TB, typename TC, bool TRA>
__global__ void k_gemm_lds(const float* __restrict__ A, const TB* __restrict__ B,
                           TC* __restrict__ C, int M, int N, int K,
                           long long sA, long long sB, long long sC)
{
    constexpr int LDB = 40;                      // halves; mult of 8 -> 16B align
    __shared__ _Float16 Bt[128 * LDB];           // B tile transposed: Bt[n*LDB + k]

    const int tid  = threadIdx.x;
    const int lane = tid & 31;
    const int wave = tid >> 5;
    const int wm   = wave >> 1;                  // 0..3  (16-row strip in block)
    const int wn   = wave & 1;                   // 0..1  (64-col strip in block)

    const int mb = M >> 6;                       // block-tiles along M
    const int bm = (int)(blockIdx.x % mb);
    const int bn = (int)(blockIdx.x / mb);

    const float* Ab = A + (long long)blockIdx.z * sA;
    const TB*    Bb = B + (long long)blockIdx.z * sB;
    TC*          Cb = C + (long long)blockIdx.z * sC;

    const int g  = lane >> 4;                    // half-wave group
    const int ml = lane & 15;
    const int rowA    = bm * 64 + wm * 16 + ml;
    const int colBase = bn * 128;

    // B staging assignment: thread -> (k row, 16-col chunk)
    const int kk = tid >> 3;                     // 0..31
    const int nl = (tid & 7) * 16;               // 0,16,...,112

    v8f acc0 = {}, acc1 = {}, acc2 = {}, acc3 = {};
    for (int k0 = 0; k0 < K; k0 += 32) {
        __syncthreads();                         // protect previous iter's reads
        // ---- stage B(32 x 128) transposed into LDS --------------------------
        {
            const long long gbase = (long long)(k0 + kk) * N + colBase + nl;
            _Float16 tmp[16];
            if constexpr (sizeof(TB) == 2) {
                v8h b0 = *(const v8h*)(Bb + gbase);
                v8h b1 = *(const v8h*)(Bb + gbase + 8);
#pragma unroll
                for (int i = 0; i < 8; ++i) { tmp[i] = b0[i]; tmp[8 + i] = b1[i]; }
            } else {
                float4 f0 = *(const float4*)(Bb + gbase);
                float4 f1 = *(const float4*)(Bb + gbase + 4);
                float4 f2 = *(const float4*)(Bb + gbase + 8);
                float4 f3 = *(const float4*)(Bb + gbase + 12);
                tmp[0]=(_Float16)f0.x;  tmp[1]=(_Float16)f0.y;  tmp[2]=(_Float16)f0.z;  tmp[3]=(_Float16)f0.w;
                tmp[4]=(_Float16)f1.x;  tmp[5]=(_Float16)f1.y;  tmp[6]=(_Float16)f1.z;  tmp[7]=(_Float16)f1.w;
                tmp[8]=(_Float16)f2.x;  tmp[9]=(_Float16)f2.y;  tmp[10]=(_Float16)f2.z; tmp[11]=(_Float16)f2.w;
                tmp[12]=(_Float16)f3.x; tmp[13]=(_Float16)f3.y; tmp[14]=(_Float16)f3.z; tmp[15]=(_Float16)f3.w;
            }
#pragma unroll
            for (int i = 0; i < 16; ++i)
                Bt[(nl + i) * LDB + kk] = tmp[i];
        }
        __syncthreads();
        // ---- A fragment: 16x32 f16 layout; K runs {8g..8g+7},{16+8g..23+8g} --
        v16h av;
        if constexpr (!TRA) {
            const float* ar = Ab + (long long)rowA * K + k0;
            float4 r0 = *(const float4*)(ar + 8 * g);
            float4 r1 = *(const float4*)(ar + 8 * g + 4);
            float4 r2 = *(const float4*)(ar + 16 + 8 * g);
            float4 r3 = *(const float4*)(ar + 16 + 8 * g + 4);
            av[0]=(_Float16)r0.x;  av[1]=(_Float16)r0.y;  av[2]=(_Float16)r0.z;  av[3]=(_Float16)r0.w;
            av[4]=(_Float16)r1.x;  av[5]=(_Float16)r1.y;  av[6]=(_Float16)r1.z;  av[7]=(_Float16)r1.w;
            av[8]=(_Float16)r2.x;  av[9]=(_Float16)r2.y;  av[10]=(_Float16)r2.z; av[11]=(_Float16)r2.w;
            av[12]=(_Float16)r3.x; av[13]=(_Float16)r3.y; av[14]=(_Float16)r3.z; av[15]=(_Float16)r3.w;
        } else {
#pragma unroll
            for (int h = 0; h < 16; ++h) {
                const int ka = (h & 7) + ((h & 8) << 1) + 8 * g;
                av[h] = (_Float16)Ab[(long long)(k0 + ka) * M + rowA];
            }
        }
        // ---- 4 wmma from LDS (lane's B fragment: K {8g..8g+7},{16+8g..23+8g}) ------
#pragma unroll
        for (int j = 0; j < 4; ++j) {
            const int colLocal = wn * 64 + j * 16 + ml;
            const v8h* p = (const v8h*)&Bt[colLocal * LDB + 8 * g];
            v8h b0 = p[0], b1 = p[2];
            v16h bv;
#pragma unroll
            for (int i = 0; i < 8; ++i) { bv[i] = b0[i]; bv[8 + i] = b1[i]; }
            v8f* accp = (j == 0) ? &acc0 : (j == 1) ? &acc1 : (j == 2) ? &acc2 : &acc3;
            *accp = __builtin_amdgcn_wmma_f32_16x16x32_f16(
                        false, av, false, bv, (short)0, *accp, false, false);
        }
    }
#pragma unroll
    for (int j = 0; j < 4; ++j) {
        const v8f& acc = (j == 0) ? acc0 : (j == 1) ? acc1 : (j == 2) ? acc2 : acc3;
        const int col = colBase + wn * 64 + j * 16 + ml;
#pragma unroll
        for (int r = 0; r < 8; ++r) {
            const int row = bm * 64 + wm * 16 + r + 8 * g;
            Cb[(long long)row * N + col] = (TC)acc[r];
        }
    }
}

// ================= fallback strip GEMM (small M, e.g. fc6/fc7) ==============
// TRB: B stored as raw[n*K + k] (raw N x K row-major).  M%16==0, N%64==0.
template<typename TB, typename TC, bool TRB>
__global__ void k_wmma_gemm(const float* __restrict__ A, const TB* __restrict__ B,
                            TC* __restrict__ C, int M, int N, int K,
                            long long sA, long long sB, long long sC)
{
    const int lane = threadIdx.x & 31;
    const int wave = threadIdx.x >> 5;
    const int tiles_m = M >> 4;
    const int tiles_n = N >> 6;
    long long tile = (long long)blockIdx.x * (blockDim.x >> 5) + wave;
    if (tile >= (long long)tiles_m * tiles_n) return;
    const int tm = (int)(tile % tiles_m);
    const int tn = (int)(tile / tiles_m);

    const float* Ab = A + (long long)blockIdx.z * sA;
    const TB*    Bb = B + (long long)blockIdx.z * sB;
    TC*          Cb = C + (long long)blockIdx.z * sC;

    const int g  = lane >> 4;
    const int ml = lane & 15;
    const int rowA = tm * 16 + ml;

    v8f acc0 = {}, acc1 = {}, acc2 = {}, acc3 = {};
    for (int k0 = 0; k0 < K; k0 += 32) {
        v16h av;
        const float* ar = Ab + (long long)rowA * K + k0;
        float4 r0 = *(const float4*)(ar + 16 * g);
        float4 r1 = *(const float4*)(ar + 16 * g + 4);
        float4 r2 = *(const float4*)(ar + 16 * g + 8);
        float4 r3 = *(const float4*)(ar + 16 * g + 12);
        av[0]=(_Float16)r0.x;  av[1]=(_Float16)r0.y;  av[2]=(_Float16)r0.z;  av[3]=(_Float16)r0.w;
        av[4]=(_Float16)r1.x;  av[5]=(_Float16)r1.y;  av[6]=(_Float16)r1.z;  av[7]=(_Float16)r1.w;
        av[8]=(_Float16)r2.x;  av[9]=(_Float16)r2.y;  av[10]=(_Float16)r2.z; av[11]=(_Float16)r2.w;
        av[12]=(_Float16)r3.x; av[13]=(_Float16)r3.y; av[14]=(_Float16)r3.z; av[15]=(_Float16)r3.w;
#pragma unroll
        for (int j = 0; j < 4; ++j) {
            const int colB = tn * 64 + j * 16 + ml;
            v16h bv;
            if constexpr (TRB) {
                const TB* br = Bb + (long long)colB * K + k0 + 16 * g;
#pragma unroll
                for (int h = 0; h < 16; ++h) bv[h] = (_Float16)br[h];
            } else {
#pragma unroll
                for (int h = 0; h < 16; ++h)
                    bv[h] = (_Float16)Bb[(long long)(k0 + h + 16 * g) * N + colB];
            }
            v8f* accp = (j == 0) ? &acc0 : (j == 1) ? &acc1 : (j == 2) ? &acc2 : &acc3;
            *accp = __builtin_amdgcn_wmma_f32_16x16x32_f16(
                        false, av, false, bv, (short)0, *accp, false, false);
        }
    }
#pragma unroll
    for (int j = 0; j < 4; ++j) {
        const v8f& acc = (j == 0) ? acc0 : (j == 1) ? acc1 : (j == 2) ? acc2 : acc3;
        const int col = tn * 64 + j * 16 + ml;
#pragma unroll
        for (int r = 0; r < 8; ++r) {
            const int row = tm * 16 + r + 8 * g;
            Cb[(long long)row * N + col] = (TC)acc[r];
        }
    }
}

// host dispatch helpers
template<typename TB, typename TC, bool TRA>
static void gemmL(hipStream_t st, const void* A, const void* B, void* C,
                  int M, int N, int K, int batch,
                  long long sA, long long sB, long long sC)
{
    int blocks = (M / 64) * (N / 128);
    k_gemm_lds<TB, TC, TRA><<<dim3(blocks, 1, batch), 256, 0, st>>>(
        (const float*)A, (const TB*)B, (TC*)C, M, N, K, sA, sB, sC);
}
template<typename TB, typename TC, bool TRB>
static void gemmS(hipStream_t st, const void* A, const void* B, void* C,
                  int M, int N, int K, int batch,
                  long long sA, long long sB, long long sC)
{
    long long tiles = (long long)(M / 16) * (N / 64);
    int blocks = (int)((tiles + 7) / 8);
    k_wmma_gemm<TB, TC, TRB><<<dim3(blocks, 1, batch), 256, 0, st>>>(
        (const float*)A, (const TB*)B, (TC*)C, M, N, K, sA, sB, sC);
}

// ============================ pointwise / reductions ========================

__global__ void k_conv1(const float* __restrict__ pts, const float* __restrict__ W,
                        _Float16* __restrict__ out, int N)
{
    long long i = (long long)blockIdx.x * blockDim.x + threadIdx.x;
    long long total = (long long)B_ * 64 * N;
    if (i >= total) return;
    int n = (int)(i % N); long long t = i / N;
    int o = (int)(t % 64); int b = (int)(t / 64);
    const float* p = pts + ((long long)b * N + n) * 3;
    float s = W[o*3+0]*p[0] + W[o*3+1]*p[1] + W[o*3+2]*p[2];
    out[i] = (_Float16)s;
}

template<typename T>
__global__ void k_bn_stats(const T* __restrict__ x, int Bn, int C, long long N,
                           float* __restrict__ mean, float* __restrict__ rstd)
{
    int c = blockIdx.x;
    __shared__ float s1[256], s2[256];
    float sum = 0.f, sq = 0.f;
    long long cnt = (long long)Bn * N;
    for (long long t = threadIdx.x; t < cnt; t += blockDim.x) {
        long long b = t / N, n = t % N;
        float v = (float)x[(b * C + c) * N + n];
        sum += v; sq += v * v;
    }
    s1[threadIdx.x] = sum; s2[threadIdx.x] = sq; __syncthreads();
    for (int s = 128; s > 0; s >>= 1) {
        if ((int)threadIdx.x < s) { s1[threadIdx.x] += s1[threadIdx.x+s]; s2[threadIdx.x] += s2[threadIdx.x+s]; }
        __syncthreads();
    }
    if (threadIdx.x == 0) {
        float m = s1[0] / (float)cnt;
        float v = s2[0] / (float)cnt - m * m;
        mean[c] = m; rstd[c] = rsqrtf(v + 1e-5f);
    }
}

template<typename Tin, typename Tout>
__global__ void k_bn_apply(const Tin* __restrict__ x, Tout* __restrict__ y,
                           const float* __restrict__ mean, const float* __restrict__ rstd,
                           const float* __restrict__ g, const float* __restrict__ beta,
                           int C, long long N, long long total, float slope)
{
    long long i = (long long)blockIdx.x * blockDim.x + threadIdx.x;
    if (i >= total) return;
    int c = (int)((i / N) % C);
    float v = ((float)x[i] - mean[c]) * rstd[c] * g[c] + beta[c];
    v = v >= 0.f ? v : v * slope;
    y[i] = (Tout)v;
}

__global__ void k_fps(const float* __restrict__ xyz, int N, int npoint, int* __restrict__ out)
{
    int b = blockIdx.x;
    const float* X = xyz + (long long)b * N * 3;
    float ld[16];
    int cnt = N / 256;
    for (int i = 0; i < cnt; ++i) ld[i] = 1e10f;
    __shared__ float rv[256]; __shared__ int ri[256]; __shared__ int sfar;
    if (threadIdx.x == 0) sfar = 0;
    __syncthreads();
    for (int j = 0; j < npoint; ++j) {
        int far = sfar;
        if (threadIdx.x == 0) out[b * npoint + j] = far;
        float cx = X[far*3], cy = X[far*3+1], cz = X[far*3+2];
        float best = -1e30f; int bi = threadIdx.x;
        for (int i = 0; i < cnt; ++i) {
            int n = threadIdx.x + i * 256;
            float dx = X[n*3]-cx, dy = X[n*3+1]-cy, dz = X[n*3+2]-cz;
            float dd = dx*dx + dy*dy + dz*dz;
            if (dd < ld[i]) ld[i] = dd;
            if (ld[i] > best) { best = ld[i]; bi = n; }
        }
        rv[threadIdx.x] = best; ri[threadIdx.x] = bi;
        __syncthreads();
        for (int s = 128; s > 0; s >>= 1) {
            if ((int)threadIdx.x < s) {
                if (rv[threadIdx.x+s] > rv[threadIdx.x] ||
                    (rv[threadIdx.x+s] == rv[threadIdx.x] && ri[threadIdx.x+s] < ri[threadIdx.x])) {
                    rv[threadIdx.x] = rv[threadIdx.x+s]; ri[threadIdx.x] = ri[threadIdx.x+s];
                }
            }
            __syncthreads();
        }
        if (threadIdx.x == 0) sfar = ri[0];
        __syncthreads();
    }
}

__global__ void k_gather_xyz(const float* __restrict__ xyz, const int* __restrict__ idx,
                             float* __restrict__ out, int N, int S)
{
    int i = blockIdx.x * blockDim.x + threadIdx.x;
    if (i >= B_ * S) return;
    int b = i / S; int id = idx[i];
    const float* src = xyz + ((long long)b * N + id) * 3;
    float* d = out + (long long)i * 3;
    d[0] = src[0]; d[1] = src[1]; d[2] = src[2];
}

template<typename T>
__global__ void k_gather_feat(const T* __restrict__ src, const int* __restrict__ idx,
                              float* __restrict__ out, int C, int N, int S)
{
    long long i = (long long)blockIdx.x * blockDim.x + threadIdx.x;
    long long total = (long long)B_ * S * C;
    if (i >= total) return;
    int c = (int)(i % C); long long t = i / C;
    int s = (int)(t % S); int b = (int)(t / S);
    int id = idx[b * S + s];
    out[i] = (float)src[((long long)b * C + c) * N + id];
}

__global__ void k_knn(const float* __restrict__ xyz, const float* __restrict__ q,
                      int* __restrict__ out, int N, int S, int Ksel)
{
    int b = blockIdx.y, s = blockIdx.x;
    __shared__ float d[4096];
    __shared__ float rv[256]; __shared__ int ri[256];
    const float* X = xyz + (long long)b * N * 3;
    const float* Q = q + ((long long)b * S + s) * 3;
    float cx = Q[0], cy = Q[1], cz = Q[2];
    for (int n = threadIdx.x; n < N; n += 256) {
        float dx = X[n*3]-cx, dy = X[n*3+1]-cy, dz = X[n*3+2]-cz;
        d[n] = dx*dx + dy*dy + dz*dz;
    }
    __syncthreads();
    for (int j = 0; j < Ksel; ++j) {
        float best = 3.4e38f; int bi = 0;
        for (int n = threadIdx.x; n < N; n += 256)
            if (d[n] < best) { best = d[n]; bi = n; }
        rv[threadIdx.x] = best; ri[threadIdx.x] = bi;
        __syncthreads();
        for (int st = 128; st > 0; st >>= 1) {
            if ((int)threadIdx.x < st) {
                if (rv[threadIdx.x+st] < rv[threadIdx.x] ||
                    (rv[threadIdx.x+st] == rv[threadIdx.x] && ri[threadIdx.x+st] < ri[threadIdx.x])) {
                    rv[threadIdx.x] = rv[threadIdx.x+st]; ri[threadIdx.x] = ri[threadIdx.x+st];
                }
            }
            __syncthreads();
        }
        if (threadIdx.x == 0) { out[((long long)b * S + s) * Ksel + j] = ri[0]; d[ri[0]] = 3.4e38f; }
        __syncthreads();
    }
}

template<typename T>
__global__ void k_build_X(const T* __restrict__ feat, const float* __restrict__ newp,
                          const int* __restrict__ knn, _Float16* __restrict__ X,
                          int Cin, int Nprev, int S, int K)
{
    long long L = (long long)B_ * S * K;
    long long l = (long long)blockIdx.x * blockDim.x + threadIdx.x;
    int dch = blockIdx.y;
    if (l >= L) return;
    int k = (int)(l % K); long long t = l / K;
    int s = (int)(t % S); int b = (int)(t / S);
    float v;
    if (dch < Cin) {
        int id = knn[((long long)b * S + s) * K + k];
        v = (float)feat[((long long)b * Cin + dch) * Nprev + id]
            - newp[((long long)b * S + s) * Cin + dch];
    } else {
        v = newp[((long long)b * S + s) * Cin + (dch - Cin)];
    }
    X[(long long)dch * L + l] = (_Float16)v;
}

__global__ void k_mean_k(const _Float16* __restrict__ H, float* __restrict__ out,
                         int C, int S, int K)
{
    long long i = (long long)blockIdx.x * blockDim.x + threadIdx.x;
    long long total = (long long)B_ * C * S;
    if (i >= total) return;
    int s = (int)(i % S); long long t = i / S;
    int c = (int)(t % C); int b = (int)(t / C);
    long long L = (long long)B_ * S * K;
    long long base = (long long)c * L + ((long long)b * S + s) * K;
    float sum = 0.f;
    for (int k = 0; k < K; ++k) sum += (float)H[base + k];
    out[i] = sum / (float)K;
}

__global__ void k_pos(const float* __restrict__ nxyz, const float* __restrict__ W,
                      const float* __restrict__ bias, float* __restrict__ pos, int S)
{
    long long i = (long long)blockIdx.x * blockDim.x + threadIdx.x;
    long long total = (long long)B_ * 256 * S;
    if (i >= total) return;
    int n = (int)(i % S); long long t = i / S;
    int c = (int)(t % 256); int b = (int)(t / 256);
    const float* p = nxyz + ((long long)b * S + n) * 3;
    pos[i] = W[c*3]*p[0] + W[c*3+1]*p[1] + W[c*3+2]*p[2] + bias[c];
}

__global__ void k_add(const float* a, const float* b, float* o, long long n)
{ long long i=(long long)blockIdx.x*blockDim.x+threadIdx.x; if(i<n) o[i]=a[i]+b[i]; }
__global__ void k_sub(const float* a, const float* b, float* o, long long n)
{ long long i=(long long)blockIdx.x*blockDim.x+threadIdx.x; if(i<n) o[i]=a[i]-b[i]; }

__global__ void k_softmax(float* __restrict__ att, int M)
{
    long long row = blockIdx.x;
    float* r = att + row * M;
    __shared__ float red[256];
    int t = threadIdx.x;
    float v = (t < M) ? r[t] : -3.4e38f;
    red[t] = v; __syncthreads();
    for (int s = 128; s > 0; s >>= 1) { if (t < s) red[t] = fmaxf(red[t], red[t+s]); __syncthreads(); }
    float mx = red[0]; __syncthreads();
    float e = (t < M) ? __expf(v - mx) : 0.f;
    red[t] = e; __syncthreads();
    for (int s = 128; s > 0; s >>= 1) { if (t < s) red[t] += red[t+s]; __syncthreads(); }
    float sum = red[0];
    if (t < M) r[t] = e / sum;
}

__global__ void k_colnorm(float* __restrict__ att, int Nrow, int M)
{
    int i = blockIdx.x * blockDim.x + threadIdx.x;
    if (i >= B_ * M) return;
    int m = i % M, b = i / M;
    float* base = att + (long long)b * Nrow * M + m;
    float s = 0.f;
    for (int n = 0; n < Nrow; ++n) s += base[(long long)n * M];
    float inv = 1.f / (1e-9f + s);
    for (int n = 0; n < Nrow; ++n) base[(long long)n * M] *= inv;
}

__global__ void k_cat_rows(const float* __restrict__ src, float* __restrict__ dst,
                           int C, int CT, int S, int co)
{
    long long i = (long long)blockIdx.x * blockDim.x + threadIdx.x;
    long long total = (long long)B_ * C * S;
    if (i >= total) return;
    int n = (int)(i % S); long long t = i / S;
    int c = (int)(t % C); int b = (int)(t / C);
    dst[((long long)b * CT + co + c) * S + n] = src[i];
}

__global__ void k_mean_n(const float* __restrict__ x, float* __restrict__ out, int C, int N)
{
    int i = blockIdx.x * blockDim.x + threadIdx.x;
    if (i >= B_ * C) return;
    const float* r = x + (long long)i * N;
    float s = 0.f;
    for (int n = 0; n < N; ++n) s += r[n];
    out[i] = s / (float)N;
}

__global__ void k_fc8(const float* __restrict__ x, const float* __restrict__ w,
                      const float* __restrict__ b, float* __restrict__ out)
{
    int i = threadIdx.x;
    if (i >= B_) return;
    float s = 0.f;
    for (int c = 0; c < 64; ++c) s += x[i*64 + c] * w[c];
    out[i] = s + b[0];
}

// =============================== driver =====================================
static inline int nb(long long total) { return (int)((total + 255) / 256); }

extern "C" void kernel_launch(void* const* d_in, const int* in_sizes, int n_in,
                              void* d_out, int out_size, void* d_ws, size_t ws_size,
                              hipStream_t stream)
{
    (void)in_sizes; (void)n_in; (void)out_size; (void)ws_size;
    auto F = [&](int i) { return (const float*)d_in[i]; };

    const float* points  = F(0);
    const float* conv1_w = F(1);  const float* bn1_g = F(2);  const float* bn1_b = F(3);
    const float* conv2_w = F(4);  const float* bn2_g = F(5);  const float* bn2_b = F(6);
    const float* lo0_w1 = F(7);  const float* lo0_g1 = F(8);  const float* lo0_b1 = F(9);
    const float* lo0_w2 = F(10); const float* lo0_g2 = F(11); const float* lo0_b2 = F(12);
    const float* lo1_w1 = F(13); const float* lo1_g1 = F(14); const float* lo1_b1 = F(15);
    const float* lo1_w2 = F(16); const float* lo1_g2 = F(17); const float* lo1_b2 = F(18);
    const float* c1w  = F(19); const float* trg = F(20); const float* trb = F(21);
    const float* posw = F(22); const float* posb = F(23);
    // sa[li]: qw,kw,vw,zw,zb,zg,zbeta at 24+7*li   (zb cancels under BN)
    const float* fuse_w = F(52); const float* fuse_g = F(53); const float* fuse_b = F(54);
    const float* fc6_w = F(55); const float* bn6_g = F(57); const float* bn6_b = F(58);
    const float* fc7_w = F(59); const float* bn7_g = F(61); const float* bn7_b = F(62);
    const float* fc8_w = F(63); const float* fc8_b = F(64);

    // ---- workspace layout (deterministic bump allocator) ----
    size_t off = 0;
    auto alloc = [&](size_t bytes) -> char* {
        char* p = (char*)d_ws + off;
        off += (bytes + 255) & ~(size_t)255;
        return p;
    };
    _Float16* P  = (_Float16*)alloc((size_t)B_*64*4096*2);
    _Float16* Q  = (_Float16*)alloc((size_t)B_*64*4096*2);
    float* meanb = (float*)alloc(2048*4);
    float* rstdb = (float*)alloc(2048*4);
    int*   fps1  = (int*)  alloc((size_t)B_*512*4);
    float* nx1   = (float*)alloc((size_t)B_*512*3*4);
    float* np1   = (float*)alloc((size_t)B_*512*64*4);
    int*   knn1  = (int*)  alloc((size_t)B_*512*32*4);
    float* f0    = (float*)alloc((size_t)B_*128*512*4);
    int*   fps2  = (int*)  alloc((size_t)B_*256*4);
    float* nx2   = (float*)alloc((size_t)B_*256*3*4);
    float* np2   = (float*)alloc((size_t)B_*256*128*4);
    int*   knn2  = (int*)  alloc((size_t)B_*256*32*4);
    float* f1    = (float*)alloc((size_t)B_*256*256*4);
    float* xv    = (float*)alloc((size_t)B_*1024*4);
    float* o6    = (float*)alloc((size_t)B_*512*4);
    float* o7    = (float*)alloc((size_t)B_*64*4);
    const size_t XELEM = 33554432;                           // 128*262144 == 256*131072
    _Float16* XA = (_Float16*)alloc(XELEM*2);
    _Float16* XB = (_Float16*)alloc(XELEM*2);

    // transformer buffers overlay XA/XB (both free once f1 is produced)
    char* ovl = (char*)XA;
    const long long TT = (long long)B_*256*256;
    float* pos  = (float*)ovl; ovl += TT*4;
    float* xb   = (float*)ovl; ovl += TT*4;
    float* qb   = (float*)ovl; ovl += (long long)B_*64*256*4;
    float* kb   = (float*)ovl; ovl += (long long)B_*64*256*4;
    float* vb   = (float*)ovl; ovl += TT*4;
    float* attb = (float*)ovl; ovl += TT*4;
    float* zbuf = (float*)ovl; ovl += TT*4;
    float* rb   = (float*)ovl; ovl += TT*4;
    float* zzb  = (float*)ovl; ovl += TT*4;
    char* ovl2 = (char*)XB;
    float* cat  = (float*)ovl2; ovl2 += (long long)B_*1280*256*4;
    float* fout = (float*)ovl2;

    // ================= Stage A: conv1 -> bn -> relu -> conv2 -> bn -> relu ==
    long long tA = (long long)B_*64*4096;
    k_conv1<<<nb(tA), 256, 0, stream>>>(points, conv1_w, P, 4096);
    k_bn_stats<_Float16><<<64, 256, 0, stream>>>(P, B_, 64, 4096, meanb, rstdb);
    k_bn_apply<_Float16,_Float16><<<nb(tA), 256, 0, stream>>>(P, P, meanb, rstdb, bn1_g, bn1_b, 64, 4096, tA, 0.f);
    gemmL<_Float16,_Float16,false>(stream, conv2_w, P, Q, 64, 4096, 64, B_,
                                   0, 64LL*4096, 64LL*4096);
    k_bn_stats<_Float16><<<64, 256, 0, stream>>>(Q, B_, 64, 4096, meanb, rstdb);
    k_bn_apply<_Float16,_Float16><<<nb(tA), 256, 0, stream>>>(Q, Q, meanb, rstdb, bn2_g, bn2_b, 64, 4096, tA, 0.f);

    // ================= Stage B1: sample_and_group(512,32) + local_op lo0 ====
    k_fps<<<B_, 256, 0, stream>>>(points, 4096, 512, fps1);
    k_gather_xyz<<<nb((long long)B_*512), 256, 0, stream>>>(points, fps1, nx1, 4096, 512);
    k_gather_feat<_Float16><<<nb((long long)B_*512*64), 256, 0, stream>>>(Q, fps1, np1, 64, 4096, 512);
    k_knn<<<dim3(512, B_), 256, 0, stream>>>(points, nx1, knn1, 4096, 512, 32);

    const long long L1 = (long long)B_*512*32;               // 262144
    k_build_X<_Float16><<<dim3((unsigned)(L1/256), 128), 256, 0, stream>>>(Q, np1, knn1, XA, 64, 4096, 512, 32);
    gemmL<_Float16,_Float16,false>(stream, lo0_w1, XA, XB, 128, (int)L1, 128, 1, 0, 0, 0);
    k_bn_stats<_Float16><<<128, 256, 0, stream>>>(XB, 1, 128, L1, meanb, rstdb);
    k_bn_apply<_Float16,_Float16><<<nb(128*L1), 256, 0, stream>>>(XB, XB, meanb, rstdb, lo0_g1, lo0_b1, 128, L1, 128*L1, 0.f);
    gemmL<_Float16,_Float16,false>(stream, lo0_w2, XB, XA, 128, (int)L1, 128, 1, 0, 0, 0);
    k_bn_stats<_Float16><<<128, 256, 0, stream>>>(XA, 1, 128, L1, meanb, rstdb);
    k_bn_apply<_Float16,_Float16><<<nb(128*L1), 256, 0, stream>>>(XA, XA, meanb, rstdb, lo0_g2, lo0_b2, 128, L1, 128*L1, 0.f);
    k_mean_k<<<nb((long long)B_*128*512), 256, 0, stream>>>(XA, f0, 128, 512, 32);

    // ================= Stage B2: sample_and_group(256,32) + local_op lo1 ====
    k_fps<<<B_, 256, 0, stream>>>(nx1, 512, 256, fps2);
    k_gather_xyz<<<nb((long long)B_*256), 256, 0, stream>>>(nx1, fps2, nx2, 512, 256);
    k_gather_feat<float><<<nb((long long)B_*256*128), 256, 0, stream>>>(f0, fps2, np2, 128, 512, 256);
    k_knn<<<dim3(256, B_), 256, 0, stream>>>(nx1, nx2, knn2, 512, 256, 32);

    const long long L2 = (long long)B_*256*32;               // 131072
    k_build_X<float><<<dim3((unsigned)(L2/256), 256), 256, 0, stream>>>(f0, np2, knn2, XA, 128, 512, 256, 32);
    gemmL<_Float16,_Float16,false>(stream, lo1_w1, XA, XB, 256, (int)L2, 256, 1, 0, 0, 0);
    k_bn_stats<_Float16><<<256, 256, 0, stream>>>(XB, 1, 256, L2, meanb, rstdb);
    k_bn_apply<_Float16,_Float16><<<nb(256*L2), 256, 0, stream>>>(XB, XB, meanb, rstdb, lo1_g1, lo1_b1, 256, L2, 256*L2, 0.f);
    gemmL<_Float16,_Float16,false>(stream, lo1_w2, XB, XA, 256, (int)L2, 256, 1, 0, 0, 0);
    k_bn_stats<_Float16><<<256, 256, 0, stream>>>(XA, 1, 256, L2, meanb, rstdb);
    k_bn_apply<_Float16,_Float16><<<nb(256*L2), 256, 0, stream>>>(XA, XA, meanb, rstdb, lo1_g2, lo1_b2, 256, L2, 256*L2, 0.f);
    k_mean_k<<<nb((long long)B_*256*256), 256, 0, stream>>>(XA, f1, 256, 256, 32);
    // XA/XB region now free -> transformer overlay is live from here on.

    // ================= Stage C: point transformer (4 SA layers) =============
    k_pos<<<nb(TT), 256, 0, stream>>>(nx2, posw, posb, pos, 256);
    gemmL<float,float,false>(stream, c1w, f1, xb, 256, 256, 256, B_, 0, 65536, 65536);
    k_bn_stats<float><<<256, 256, 0, stream>>>(xb, B_, 256, 256, meanb, rstdb);
    k_bn_apply<float,float><<<nb(TT), 256, 0, stream>>>(xb, xb, meanb, rstdb, trg, trb, 256, 256, TT, 0.f);

    for (int li = 0; li < 4; ++li) {
        const float* qw = F(24 + 7*li); const float* kw = F(25 + 7*li);
        const float* vw = F(26 + 7*li); const float* zw = F(27 + 7*li);
        const float* zg = F(29 + 7*li); const float* zbeta = F(30 + 7*li);

        k_add<<<nb(TT), 256, 0, stream>>>(xb, pos, xb, TT);                     // x += pos
        gemmL<float,float,false>(stream, qw, xb, qb, 64, 256, 256, B_, 0, 65536, 16384);
        gemmL<float,float,false>(stream, kw, xb, kb, 64, 256, 256, B_, 0, 65536, 16384);
        gemmL<float,float,false>(stream, vw, xb, vb, 256, 256, 256, B_, 0, 65536, 65536);
        // att[n,m] = sum_c q[c,n] k[c,m]  -> A = q^T (transA)
        gemmL<float,float,true>(stream, qb, kb, attb, 256, 256, 64, B_, 16384, 16384, 65536);
        k_softmax<<<B_*256, 256, 0, stream>>>(attb, 256);
        k_colnorm<<<nb(B_*256), 256, 0, stream>>>(attb, 256, 256);
        gemmL<float,float,false>(stream, vb, attb, zbuf, 256, 256, 256, B_, 65536, 65536, 65536);
        k_sub<<<nb(TT), 256, 0, stream>>>(xb, zbuf, rb, TT);                    // r = x - z
        gemmL<float,float,false>(stream, zw, rb, zzb, 256, 256, 256, B_, 0, 65536, 65536);
        // zb bias cancels inside bn_cn
        k_bn_stats<float><<<256, 256, 0, stream>>>(zzb, B_, 256, 256, meanb, rstdb);
        k_bn_apply<float,float><<<nb(TT), 256, 0, stream>>>(zzb, zzb, meanb, rstdb, zg, zbeta, 256, 256, TT, 0.f);
        k_add<<<nb(TT), 256, 0, stream>>>(xb, zzb, xb, TT);                     // x = x + relu(bn(...))
        k_cat_rows<<<nb(TT), 256, 0, stream>>>(xb, cat, 256, 1280, 256, 256 * li);
    }
    k_cat_rows<<<nb(TT), 256, 0, stream>>>(f1, cat, 256, 1280, 256, 1024);

    // ================= Stage D: fuse -> mean -> fc6/fc7/fc8 =================
    gemmL<float,float,false>(stream, fuse_w, cat, fout, 1024, 256, 1280, B_,
                             0, 1280LL*256, 1024LL*256);
    long long tF = (long long)B_*1024*256;
    k_bn_stats<float><<<1024, 256, 0, stream>>>(fout, B_, 1024, 256, meanb, rstdb);
    k_bn_apply<float,float><<<nb(tF), 256, 0, stream>>>(fout, fout, meanb, rstdb, fuse_g, fuse_b, 1024, 256, tF, 0.2f);
    k_mean_n<<<nb(B_*1024), 256, 0, stream>>>(fout, xv, 1024, 256);

    gemmS<float,float,true>(stream, xv, fc6_w, o6, 16, 512, 1024, 1, 0, 0, 0);   // fc6_b cancels in bn_c
    k_bn_stats<float><<<512, 256, 0, stream>>>(o6, B_, 512, 1, meanb, rstdb);
    k_bn_apply<float,float><<<nb(B_*512), 256, 0, stream>>>(o6, o6, meanb, rstdb, bn6_g, bn6_b, 512, 1, B_*512, 0.01f);

    gemmS<float,float,true>(stream, o6, fc7_w, o7, 16, 64, 512, 1, 0, 0, 0);     // fc7_b cancels in bn_c
    k_bn_stats<float><<<64, 256, 0, stream>>>(o7, B_, 64, 1, meanb, rstdb);
    k_bn_apply<float,float><<<nb(B_*64), 256, 0, stream>>>(o7, o7, meanb, rstdb, bn7_g, bn7_b, 64, 1, B_*64, 0.01f);

    k_fc8<<<1, 32, 0, stream>>>(o7, fc8_w, fc8_b, (float*)d_out);
}
